// MyModel_61933428409343
// MI455X (gfx1250) — matrix operand order repaired and checked by reading, not compiled, not run on hardware
//
#include <hip/hip_runtime.h>
#include <hip/hip_bf16.h>

#define IN_DIM 10
#define HDIM   20
#define BATCH  2048
#define TSTEPS 512
#define NPAD   96   // 3 gates, each padded 20 -> 32 columns
#define KPAD   32   // K padded to 32 for wmma 16x16x32

typedef __attribute__((ext_vector_type(16))) _Float16 v16h;
typedef __attribute__((ext_vector_type(8)))  _Float16 v8h;
typedef __attribute__((ext_vector_type(8)))  float    v8f;

__device__ inline v8f wmma_f16(v16h a, v16h b, v8f c) {
    return __builtin_amdgcn_wmma_f32_16x16x32_f16(
        /*neg_a=*/false, a, /*neg_b=*/false, b,
        /*c_mod=*/(short)0, c, /*reuse_a=*/false, /*reuse_b=*/false);
}

// A-fragment (16x32 f16): lane m=lane&15; lanes 0-15 hold K = 0..7 & 16..23,
// lanes 16-31 hold K = 8..15 & 24..31 (interleaved per ISA table).
__device__ inline v16h afrag_row(const _Float16* row, int hi) {
    const v8h* p = (const v8h*)row;
    v8h c1 = p[hi];        // k = hi*8 .. hi*8+7
    v8h c2 = p[2 + hi];    // k = 16+hi*8 ..
    return __builtin_shufflevector(c1, c2, 0,1,2,3,4,5,6,7,8,9,10,11,12,13,14,15);
}

// B-fragment (32x16 f16): lane n=lane&15; lanes 0-15 hold K = 0..7 & 16..23,
// lanes 16-31 hold K = 8..15 & 24..31 (same interleave as A-fragment).
__device__ inline v16h bfrag_row(const _Float16* row, int hi) {
    const v8h* p = (const v8h*)row;
    v8h c1 = p[hi];
    v8h c2 = p[2 + hi];
    return __builtin_shufflevector(c1, c2, 0,1,2,3,4,5,6,7,8,9,10,11,12,13,14,15);
}

// Fast activations: single v_exp_f32 + single v_rcp_f32 (no IEEE div expansion).
__device__ inline float fast_sigmoid(float x) {
    return __builtin_amdgcn_rcpf(1.0f + __expf(-x));
}
__device__ inline float fast_tanh(float x) {
#if __has_builtin(__builtin_amdgcn_tanhf)
    return __builtin_amdgcn_tanhf(x);     // gfx1250 v_tanh_f32
#else
    // tanh(x) = 1 - 2/(exp(2x)+1)
    float e = __expf(2.0f * x);
    return 1.0f - 2.0f * __builtin_amdgcn_rcpf(e + 1.0f);
#endif
}

__global__ __launch_bounds__(128) void gru2_fused_kernel(
    const float* __restrict__ x,
    const float* __restrict__ w_ih0, const float* __restrict__ w_hh0,
    const float* __restrict__ b_ih0, const float* __restrict__ b_hh0,
    const float* __restrict__ w_ih1, const float* __restrict__ w_hh1,
    const float* __restrict__ b_ih1, const float* __restrict__ b_hh1,
    float* __restrict__ out)
{
    // Weight buffers in B-fragment-friendly [n][k] layout, gate-padded to 96x32.
    __shared__ __align__(16) _Float16 wbuf[4][NPAD][KPAD]; // wih0, whh0, wih1, whh1
    // Per-wave staging for D->A layout round-trips (2 per wave: layer0, layer1)
    __shared__ __align__(16) _Float16 stage[4][2][16][KPAD];

    const int tid = threadIdx.x;

    // ---- Cooperative weight pack: f32 [60,K] -> f16 padded [96,32] ----
    {
        const float* wsrc[4] = { w_ih0, w_hh0, w_ih1, w_hh1 };
        const int    kdim[4] = { IN_DIM, HDIM, HDIM, HDIM };
        #pragma unroll
        for (int m = 0; m < 4; ++m) {
            const float* src = wsrc[m];
            const int K = kdim[m];
            for (int idx = tid; idx < NPAD * KPAD; idx += (int)blockDim.x) {
                int n = idx >> 5, k = idx & 31;
                int p = n >> 5, w = n & 31;       // gate p (r,z,n), col-in-gate w
                float val = (w < HDIM && k < K) ? src[(p*HDIM + w)*K + k] : 0.0f;
                wbuf[m][n][k] = (_Float16)val;
            }
        }
    }
    __syncthreads();

    const int wave = tid >> 5;
    const int lane = tid & 31;
    const int col  = lane & 15;   // N (or M for A-frag rows)
    const int hi   = lane >> 4;
    const int tileIdx = blockIdx.x * 4 + wave;   // 0..127 batch tiles
    const int b0 = tileIdx * 16;

    // ---- Per-lane per-tile biases (column-dependent only) ----
    float bi0[6], bh0[6], bi1[6], bh1[6];
    #pragma unroll
    for (int tN = 0; tN < 6; ++tN) {
        int n = tN*16 + col;
        int p = n >> 5, w = n & 31;
        bool valid = (w < HDIM);
        int r = p*HDIM + w;
        bi0[tN] = valid ? b_ih0[r] : 0.0f;
        bh0[tN] = valid ? b_hh0[r] : 0.0f;
        bi1[tN] = valid ? b_ih1[r] : 0.0f;
        bh1[tN] = valid ? b_hh1[r] : 0.0f;
    }

    // ---- Hidden state: A-fragment (f16, for matmul) + D-layout f32 (for update) ----
    v16h h0A = {};
    v16h h1A = {};
    v8f  h0D[2] = {};   // cols 0-15, 16-31 (valid 16-19)
    v8f  h1D[2] = {};

    // One GRU layer step: consumes input A-frag, updates hD and hA in place.
    auto layer_step = [&](v16h inA, const _Float16* wih, const _Float16* whh,
                          const float* bi, const float* bh,
                          v16h& hA, v8f* hD, _Float16 (*stg)[KPAD]) {
        v8f hnew[2];
        #pragma unroll
        for (int sub = 0; sub < 2; ++sub) {
            const int tr = sub, tz = 2 + sub, tn = 4 + sub;
            v8f zero = {};
            // r, z preactivations: x-proj chained into h-proj via C operand
            v8f accR = wmma_f16(inA, bfrag_row(wih + (tr*16 + col)*KPAD, hi), zero);
            accR     = wmma_f16(hA,  bfrag_row(whh + (tr*16 + col)*KPAD, hi), accR);
            v8f accZ = wmma_f16(inA, bfrag_row(wih + (tz*16 + col)*KPAD, hi), zero);
            accZ     = wmma_f16(hA,  bfrag_row(whh + (tz*16 + col)*KPAD, hi), accZ);
            // n gate: keep x-part and h-part separate (r multiplies h-part + b_hh)
            v8f xn = wmma_f16(inA, bfrag_row(wih + (tn*16 + col)*KPAD, hi), zero);
            v8f hn = wmma_f16(hA,  bfrag_row(whh + (tn*16 + col)*KPAD, hi), zero);

            const float bR = bi[tr] + bh[tr];
            const float bZ = bi[tz] + bh[tz];
            #pragma unroll
            for (int e = 0; e < 8; ++e) {
                float r  = fast_sigmoid(accR[e] + bR);
                float z  = fast_sigmoid(accZ[e] + bZ);
                float nn = fast_tanh(xn[e] + bi[tn] + r * (hn[e] + bh[tn]));
                hnew[sub][e] = (1.0f - z) * nn + z * hD[sub][e];
            }
        }
        hD[0] = hnew[0];
        hD[1] = hnew[1];
        // D-layout -> A-fragment round-trip through per-wave LDS staging
        #pragma unroll
        for (int v = 0; v < 8; ++v) {
            int m = v + 8*hi;
            stg[m][col]      = (_Float16)hnew[0][v];
            stg[m][16 + col] = (col < 4) ? (_Float16)hnew[1][v] : (_Float16)0.0f;
        }
        __syncthreads();
        hA = afrag_row(&stg[col][0], hi);
        // WAR on staging is covered by the other layer's barrier next call.
    };

    for (int t = 0; t < TSTEPS; ++t) {
        // Build layer-0 input A-fragment directly from x[b, t, 0:10]
        v16h xA = {};
        {
            const float* xr = x + ((size_t)(b0 + col) * TSTEPS + (size_t)t) * IN_DIM;
            if (hi == 0) {
                #pragma unroll
                for (int k = 0; k < 8; ++k) xA[k] = (_Float16)xr[k];   // K = 0..7
            } else {
                xA[0] = (_Float16)xr[8];                               // K = 8
                xA[1] = (_Float16)xr[9];                               // K = 9
            }
            // K >= 10 stays zero (weights also zero-padded)
        }

        layer_step(xA,  &wbuf[0][0][0], &wbuf[1][0][0], bi0, bh0, h0A, h0D, stage[wave][0]);
        // layer-1 input is layer-0's freshly updated hidden fragment
        layer_step(h0A, &wbuf[2][0][0], &wbuf[3][0][0], bi1, bh1, h1A, h1D, stage[wave][1]);

        // Store layer-1 hidden to out[b, t, 0:20] (f32)
        #pragma unroll
        for (int v = 0; v < 8; ++v) {
            int m = v + 8*hi;
            size_t base = ((size_t)(b0 + m) * TSTEPS + (size_t)t) * HDIM;
            out[base + col] = h1D[0][v];
            if (col < 4) out[base + 16 + col] = h1D[1][v];
        }
    }
}

extern "C" void kernel_launch(void* const* d_in, const int* in_sizes, int n_in,
                              void* d_out, int out_size, void* d_ws, size_t ws_size,
                              hipStream_t stream) {
    const float* x     = (const float*)d_in[0];
    const float* w_ih0 = (const float*)d_in[1];
    const float* w_hh0 = (const float*)d_in[2];
    const float* b_ih0 = (const float*)d_in[3];
    const float* b_hh0 = (const float*)d_in[4];
    const float* w_ih1 = (const float*)d_in[5];
    const float* w_hh1 = (const float*)d_in[6];
    const float* b_ih1 = (const float*)d_in[7];
    const float* b_hh1 = (const float*)d_in[8];
    float* out = (float*)d_out;

    // 128 batch tiles of 16 rows; 4 waves (one tile each) per workgroup.
    dim3 grid(BATCH / 16 / 4);   // 32
    dim3 block(128);
    gru2_fused_kernel<<<grid, block, 0, stream>>>(
        x, w_ih0, w_hh0, b_ih0, b_hh0, w_ih1, w_hh1, b_ih1, b_hh1, out);
}
